// GPSConvModule_17978733101126
// MI455X (gfx1250) — matrix operand order repaired and checked by reading, not compiled, not run on hardware
//
#include <hip/hip_runtime.h>
#include <math.h>
#include <stdint.h>

typedef _Float16 h16;
typedef __attribute__((ext_vector_type(16))) _Float16 v16h;
typedef __attribute__((ext_vector_type(8)))  _Float16 v8h;
typedef __attribute__((ext_vector_type(4)))  _Float16 v4h;
typedef __attribute__((ext_vector_type(8)))  float    v8f;
typedef __attribute__((ext_vector_type(4)))  unsigned int u32x4;
typedef __attribute__((ext_vector_type(8)))  int i32x8;
typedef __attribute__((ext_vector_type(4)))  int i32x4;

#define BATCH 4
#define NNODES 4096
#define FIN 128
#define CDIM 64
#define HEADS 4
#define MROWS (BATCH*NNODES)   // 16384

#if defined(__HIP_DEVICE_COMPILE__) && __has_builtin(__builtin_amdgcn_tensor_load_to_lds)
#define USE_TDM 1
#else
#define USE_TDM 0
#endif

// ---------------------------------------------------------------------------
// f32 -> f16 conversion (weights + activations fed to WMMA)
// ---------------------------------------------------------------------------
__global__ void f32_to_f16_kernel(const float* __restrict__ in, h16* __restrict__ out, int n) {
  int i = blockIdx.x * blockDim.x + threadIdx.x;
  if (i < n) out[i] = (h16)in[i];
}

// ---------------------------------------------------------------------------
// WMMA fragment helpers (gfx1250 wave32 layouts, CDNA5 ISA §7.12.2)
// A 16x32 f16: lane m=lane&15, q=lane>>4; halves = k(q*16..q*16+15) (contiguous order chosen here, not ISA table), row m
// B 32x16 f16: lane n=lane&15, q=lane>>4; halves = k(q*16..q*16+15) (contiguous order chosen here, not ISA table), col n
// C/D 16x16 f32: VGPR r, lane L: (m = r + 8*(L>=16), n = L&15)
// ---------------------------------------------------------------------------
__device__ __forceinline__ v16h a_frag_from_row(const h16* __restrict__ rowptr, int q) {
  v8h lo = *(const v8h*)(rowptr + q * 16);
  v8h hi = *(const v8h*)(rowptr + q * 16 + 8);
  v16h a;
#pragma unroll
  for (int i = 0; i < 8; ++i) { a[i] = lo[i]; a[i + 8] = hi[i]; }
  return a;
}

__device__ __forceinline__ v16h b_frag_from_w(const h16* __restrict__ W, int ldw,
                                              int n0, int k0, int lane) {
  int n = lane & 15, q = lane >> 4;
  return *(const v16h*)(W + (size_t)(n0 + n) * ldw + k0 + q * 16);
}

// ---------------------------------------------------------------------------
// Generic f16 WMMA GEMM: out[M,Nout] = A[M,K] * W[Nout,K]^T + bias (+resid)(+relu)
// One wave -> one 16x16 tile.
// ---------------------------------------------------------------------------
__global__ void gemm16_kernel(const h16* __restrict__ A, const h16* __restrict__ W,
                              const float* __restrict__ bias, const float* __restrict__ resid,
                              float* __restrict__ outf, h16* __restrict__ outh,
                              int M, int Nout, int K, int relu, int total_tiles) {
  int lane = threadIdx.x & 31;
  int wid = blockIdx.x * (blockDim.x >> 5) + (threadIdx.x >> 5);
  if (wid >= total_tiles) return;                     // wave-uniform
  int ntiles = Nout >> 4;
  int m0 = (wid / ntiles) * 16;
  int n0 = (wid % ntiles) * 16;
  int q = lane >> 4;

  const h16* arow = A + (size_t)(m0 + (lane & 15)) * K;
  v8f c = {};
  for (int k0 = 0; k0 < K; k0 += 32) {
    v16h a = a_frag_from_row(arow + k0, q);
    v16h b = b_frag_from_w(W, K, n0, k0, lane);
    c = __builtin_amdgcn_wmma_f32_16x16x32_f16(false, a, false, b, (short)0, c, false, false);
  }

  int n = lane & 15;
  float bv = bias ? bias[n0 + n] : 0.f;
#pragma unroll
  for (int r = 0; r < 8; ++r) {
    int m = m0 + r + 8 * q;
    size_t o = (size_t)m * Nout + (n0 + n);
    float v = c[r] + bv;
    if (resid) v += resid[o];
    if (relu)  v = v > 0.f ? v : 0.f;
    if (outf) outf[o] = v;
    if (outh) outh[o] = (h16)v;
  }
}

// ---------------------------------------------------------------------------
// TDM descriptor for a K tile: 32 rows x 16 halves, row stride 192 halves.
// D# packing per CDNA5 ISA.
// ---------------------------------------------------------------------------
#if USE_TDM
__device__ __forceinline__ void tdm_load_k_tile(const h16* gsrc, unsigned lds_addr) {
  unsigned long long ga = (unsigned long long)(uintptr_t)gsrc;
  u32x4 g0;
  g0[0] = 1u;                                            // count=1 (valid), user mode
  g0[1] = lds_addr;                                      // lds_addr (bytes)
  g0[2] = (unsigned)(ga & 0xFFFFFFFFu);                  // global_addr[31:0]
  g0[3] = (unsigned)((ga >> 32) & 0x01FFFFFFu)           // global_addr[56:32]
          | (2u << 30);                                  // type = 2 ("image")
  i32x8 g1;
  g1[0] = (1 << 16);            // workgroup_mask=0 (not in cluster), data_size=1 (2B)
  g1[1] = (16 << 16);           // tensor_dim0[15:0]=16 in bits 63:48
  g1[2] = (4096 << 16);         // tensor_dim0 hi=0 ; tensor_dim1[15:0]=4096
  g1[3] = (16 << 16);           // tensor_dim1 hi=0 ; tile_dim0=16
  g1[4] = 32;                   // tile_dim1=32 ; tile_dim2=0 (2-D tile)
  g1[5] = 192;                  // tensor_dim0_stride[31:0] = 192 elements
  g1[6] = 0;                    // stride hi / tensor_dim1_stride lo = 0
  g1[7] = 0;
  i32x4 gz = {0, 0, 0, 0};
#if defined(__clang_major__) && (__clang_major__ >= 23)
  i32x8 gz8 = {0, 0, 0, 0, 0, 0, 0, 0};
  __builtin_amdgcn_tensor_load_to_lds(g0, g1, gz, gz, gz8, 0);
#else
  __builtin_amdgcn_tensor_load_to_lds(g0, g1, gz, gz, 0);
#endif
}
#endif

// V chunk staged transposed ([d][j_local]) with b64 loads, 4 halves/thread.
__device__ __forceinline__ void stage_v(const h16* __restrict__ qkv, int b, int h, int j0,
                                        h16* __restrict__ vt, int tid) {
  int r = tid >> 2;            // 0..31 key row
  int d0 = (tid & 3) * 4;      // 0,4,8,12
  const h16* src = qkv + ((size_t)b * NNODES + j0 + r) * 192 + 128 + h * 16 + d0;
  v4h v = *(const v4h*)src;
#pragma unroll
  for (int i = 0; i < 4; ++i) vt[(d0 + i) * 32 + r] = v[i];
}

#if !USE_TDM
__device__ __forceinline__ void stage_k(const h16* __restrict__ qkv, int b, int h, int j0,
                                        h16* __restrict__ ks, int tid) {
  int r = tid >> 2;
  int d0 = (tid & 3) * 4;
  const h16* src = qkv + ((size_t)b * NNODES + j0 + r) * 192 + 64 + h * 16 + d0;
  *(v4h*)(ks + r * 16 + d0) = *(const v4h*)src;
}
#endif

// ---------------------------------------------------------------------------
// Flash attention: qkv f16 [B*N,192] (q|k|v, head h at col h*16 within each).
// Block = one (b,h) x 64 query rows; 4 waves x 16-row tiles; j-chunks of 32.
// K tiles DMA'd to LDS by the TDM (double buffered, one chunk ahead);
// V tiles staged transposed by ALU loads. One barrier per iteration:
//   buffer cur is read in iter jt; it is next overwritten by the prefetch
//   issued in iter jt+1, which happens after the barrier at the top of jt+1,
//   by which point every wave has finished its iter-jt reads.
// ---------------------------------------------------------------------------
__global__ void flash_attn_kernel(const h16* __restrict__ qkv, h16* __restrict__ o_out) {
  int tid = threadIdx.x;
  int lane = tid & 31, wave = tid >> 5;
  int t64 = blockIdx.x & 63;           // N/64 = 64 row groups
  int bh  = blockIdx.x >> 6;           // 16 (b,h) pairs
  int b = bh >> 2, h = bh & 3;
  int row0 = t64 * 64 + wave * 16;
  int m = lane & 15, q = lane >> 4;

  __shared__ __align__(32) h16 Ks[2][32 * 16];   // K chunk row-major [j_local][d]
  __shared__ __align__(32) h16 Vt[2][16 * 32];   // V chunk transposed [d][j_local]
  __shared__ __align__(16) h16 Ps[4][16 * 32];   // per-wave P staging [m][j_local]

  const size_t ROWS = 192;
  const h16* qbase = qkv + ((size_t)b * NNODES + row0 + m) * ROWS + h * 16;

  // Q fragment: Dh=16, pad K 16->32 with zeros in the A operand
  v16h aq;
  {
    v16h f = *(const v16h*)qbase;
#pragma unroll
    for (int i = 0; i < 16; ++i) { aq[i] = q ? (h16)0.f : f[i]; }
  }

  float mrun[8], lrun[8];
  v8f acc = {};
#pragma unroll
  for (int r = 0; r < 8; ++r) { mrun[r] = -1e30f; lrun[r] = 0.f; }

  h16* pw = &Ps[wave][0];
  const float sc = 0.25f;  // 1/sqrt(Dh)
  const int NITER = NNODES / 32;

  // prologue: stage chunk 0 into buffer 0
#if USE_TDM
  if (wave == 0)
    tdm_load_k_tile(qkv + (size_t)b * NNODES * ROWS + 64 + h * 16,
                    (unsigned)(uintptr_t)&Ks[0][0]);
#else
  stage_k(qkv, b, h, 0, Ks[0], tid);
#endif
  stage_v(qkv, b, h, 0, Vt[0], tid);

  for (int jt = 0; jt < NITER; ++jt) {
    int cur = jt & 1;
#if USE_TDM
    if (wave == 0) __builtin_amdgcn_s_wait_tensorcnt(0);
#endif
    __syncthreads();                       // chunk `cur` visible to all waves
    if (jt + 1 < NITER) {                  // prefetch next chunk into cur^1
      int j1 = (jt + 1) * 32;
#if USE_TDM
      if (wave == 0)
        tdm_load_k_tile(qkv + ((size_t)b * NNODES + j1) * ROWS + 64 + h * 16,
                        (unsigned)(uintptr_t)&Ks[cur ^ 1][0]);
#else
      stage_k(qkv, b, h, j1, Ks[cur ^ 1], tid);
#endif
      stage_v(qkv, b, h, j1, Vt[cur ^ 1], tid);
    }
    const h16* KsC = Ks[cur];
    const h16* VtC = Vt[cur];

    // scores: S[16x32] = Q * K^T as two 16x16 WMMAs (A upper-K is zero)
    v16h bk0 = *(const v16h*)(KsC + (lane & 15) * 16);
    v16h bk1 = *(const v16h*)(KsC + (16 + (lane & 15)) * 16);
    v8f z = {};
    v8f s0 = __builtin_amdgcn_wmma_f32_16x16x32_f16(false, aq, false, bk0, (short)0, z, false, false);
    v8f s1 = __builtin_amdgcn_wmma_f32_16x16x32_f16(false, aq, false, bk1, (short)0, z, false, false);

    // online softmax per row (row r+8*q lives in a fixed 16-lane half)
#pragma unroll
    for (int r = 0; r < 8; ++r) {
      float a0 = s0[r] * sc, a1 = s1[r] * sc;
      float tmax = fmaxf(a0, a1);
#pragma unroll
      for (int msk = 1; msk < 16; msk <<= 1) tmax = fmaxf(tmax, __shfl_xor(tmax, msk, 32));
      float mnew  = fmaxf(mrun[r], tmax);
      float alpha = __expf(mrun[r] - mnew);
      float p0 = __expf(a0 - mnew), p1 = __expf(a1 - mnew);
      float rs = p0 + p1;
#pragma unroll
      for (int msk = 1; msk < 16; msk <<= 1) rs += __shfl_xor(rs, msk, 32);
      lrun[r] = lrun[r] * alpha + rs;
      mrun[r] = mnew;
      acc[r] *= alpha;
      int mm = r + 8 * q, nn = lane & 15;
      pw[mm * 32 + nn]      = (h16)p0;
      pw[mm * 32 + 16 + nn] = (h16)p1;
    }

    // O += P * V  (P re-read from LDS in A layout; V^T reads as contiguous B frag)
    v16h ap;
    {
      const h16* pr = pw + (lane & 15) * 32;
      v8h lo = *(const v8h*)(pr + q * 16);
      v8h hi = *(const v8h*)(pr + q * 16 + 8);
#pragma unroll
      for (int i = 0; i < 8; ++i) { ap[i] = lo[i]; ap[i + 8] = hi[i]; }
    }
    v16h bv = *(const v16h*)(VtC + (lane & 15) * 32 + q * 16);
    acc = __builtin_amdgcn_wmma_f32_16x16x32_f16(false, ap, false, bv, (short)0, acc, false, false);
  }

  // normalize and store o (f16, [B,N,64], head slice)
  int nn = lane & 15;
#pragma unroll
  for (int r = 0; r < 8; ++r) {
    int mm = r + 8 * q;
    float v = acc[r] / lrun[r];
    o_out[((size_t)b * NNODES + row0 + mm) * CDIM + h * 16 + nn] = (h16)v;
  }
}

// ---------------------------------------------------------------------------
// BatchNorm over the node axis: one block per (b,c)
// ---------------------------------------------------------------------------
__global__ void bn_stats_kernel(const float* __restrict__ x, float* __restrict__ mean,
                                float* __restrict__ rstd) {
  int b = blockIdx.x / CDIM, c = blockIdx.x % CDIM;
  float s = 0.f, s2 = 0.f;
  for (int n = threadIdx.x; n < NNODES; n += 256) {
    float v = x[((size_t)b * NNODES + n) * CDIM + c];
    s += v; s2 += v * v;
  }
  __shared__ float sh[256], sh2[256];
  int t = threadIdx.x;
  sh[t] = s; sh2[t] = s2; __syncthreads();
  for (int o = 128; o > 0; o >>= 1) {
    if (t < o) { sh[t] += sh[t + o]; sh2[t] += sh2[t + o]; }
    __syncthreads();
  }
  if (t == 0) {
    float mu = sh[0] / (float)NNODES;
    float var = sh2[0] / (float)NNODES - mu * mu;
    mean[blockIdx.x] = mu;
    rstd[blockIdx.x] = rsqrtf(var + 1e-5f);
  }
}

__global__ void bn_apply_kernel(const float* __restrict__ x, const float* __restrict__ mean,
                                const float* __restrict__ rstd, const float* __restrict__ g,
                                const float* __restrict__ bt, float* __restrict__ of,
                                h16* __restrict__ oh, int total, int do_tanh) {
  int i = blockIdx.x * blockDim.x + threadIdx.x;
  if (i >= total) return;
  int c = i % CDIM;
  int b = i / (NNODES * CDIM);
  float v = g[c] * (x[i] - mean[b * CDIM + c]) * rstd[b * CDIM + c] + bt[c];
  if (do_tanh) v = tanhf(v);
  if (of) of[i] = v;
  if (oh) oh[i] = (h16)v;
}

// ---------------------------------------------------------------------------
extern "C" void kernel_launch(void* const* d_in, const int* in_sizes, int n_in,
                              void* d_out, int out_size, void* d_ws, size_t ws_size,
                              hipStream_t stream) {
  const float* inputs = (const float*)d_in[0];
  const float* Wp   = (const float*)d_in[2];
  const float* bp   = (const float*)d_in[3];
  const float* Wqkv = (const float*)d_in[4];
  const float* bqkv = (const float*)d_in[5];
  const float* Wo   = (const float*)d_in[6];
  const float* bo   = (const float*)d_in[7];
  const float* g2   = (const float*)d_in[8];
  const float* b2   = (const float*)d_in[9];
  const float* W1   = (const float*)d_in[10];
  const float* b1   = (const float*)d_in[11];
  const float* W2   = (const float*)d_in[12];
  const float* b2m  = (const float*)d_in[13];
  const float* g3   = (const float*)d_in[14];
  const float* b3   = (const float*)d_in[15];
  float* out = (float*)d_out;

  // workspace carve-up (256B aligned)
  char* ws = (char*)d_ws;
  size_t off = 0;
  auto carve = [&](size_t bytes) -> char* {
    char* p = ws + off;
    off += (bytes + 255) & ~(size_t)255;
    return p;
  };
  h16*  in16   = (h16*)carve((size_t)MROWS * FIN * 2);
  h16*  wp16   = (h16*)carve((size_t)CDIM * FIN * 2);
  h16*  wqkv16 = (h16*)carve((size_t)3 * CDIM * CDIM * 2);
  h16*  wo16   = (h16*)carve((size_t)CDIM * CDIM * 2);
  h16*  w116   = (h16*)carve((size_t)2 * CDIM * CDIM * 2);
  h16*  w216   = (h16*)carve((size_t)CDIM * 2 * CDIM * 2);
  float* x32   = (float*)carve((size_t)MROWS * CDIM * 4);
  h16*  x16    = (h16*)carve((size_t)MROWS * CDIM * 2);
  h16*  qkv16  = (h16*)carve((size_t)MROWS * 3 * CDIM * 2);
  h16*  o16    = (h16*)carve((size_t)MROWS * CDIM * 2);
  float* hpre  = (float*)carve((size_t)MROWS * CDIM * 4);
  float* h32   = (float*)carve((size_t)MROWS * CDIM * 4);
  h16*  h16b   = (h16*)carve((size_t)MROWS * CDIM * 2);
  h16*  t16    = (h16*)carve((size_t)MROWS * 2 * CDIM * 2);
  float* pre3  = (float*)carve((size_t)MROWS * CDIM * 4);
  float* mean2 = (float*)carve(BATCH * CDIM * 4);
  float* rstd2 = (float*)carve(BATCH * CDIM * 4);
  float* mean3 = (float*)carve(BATCH * CDIM * 4);
  float* rstd3 = (float*)carve(BATCH * CDIM * 4);
  (void)ws_size; (void)n_in; (void)in_sizes; (void)out_size;

  auto cvt = [&](const float* src, h16* dst, int n) {
    f32_to_f16_kernel<<<(n + 255) / 256, 256, 0, stream>>>(src, dst, n);
  };
  cvt(inputs, in16, MROWS * FIN);
  cvt(Wp,   wp16,   CDIM * FIN);
  cvt(Wqkv, wqkv16, 3 * CDIM * CDIM);
  cvt(Wo,   wo16,   CDIM * CDIM);
  cvt(W1,   w116,   2 * CDIM * CDIM);
  cvt(W2,   w216,   CDIM * 2 * CDIM);

  auto gemm = [&](const h16* A, const h16* W, const float* bias, const float* resid,
                  float* outf, h16* outh, int M, int Nout, int K, int relu) {
    int tiles = (M / 16) * (Nout / 16);
    gemm16_kernel<<<tiles / 4, 128, 0, stream>>>(A, W, bias, resid, outf, outh,
                                                 M, Nout, K, relu, tiles);
  };

  // x = inputs @ Wp^T + bp            (f32 for residual, f16 for next GEMM)
  gemm(in16, wp16, bp, nullptr, x32, x16, MROWS, CDIM, FIN, 0);
  // qkv = x @ Wqkv^T + bqkv           (f16, [B*N,192])
  gemm(x16, wqkv16, bqkv, nullptr, nullptr, qkv16, MROWS, 3 * CDIM, CDIM, 0);
  // flash attention -> o (f16)
  flash_attn_kernel<<<BATCH * HEADS * (NNODES / 64), 128, 0, stream>>>(qkv16, o16);
  // h_pre = o @ Wo^T + bo + x
  gemm(o16, wo16, bo, x32, hpre, nullptr, MROWS, CDIM, CDIM, 0);
  // norm2
  bn_stats_kernel<<<BATCH * CDIM, 256, 0, stream>>>(hpre, mean2, rstd2);
  bn_apply_kernel<<<(MROWS * CDIM + 255) / 256, 256, 0, stream>>>(
      hpre, mean2, rstd2, g2, b2, h32, h16b, MROWS * CDIM, 0);
  // mlp: t = relu(h @ W1^T + b1); pre3 = t @ W2^T + b2m + h
  gemm(h16b, w116, b1, nullptr, nullptr, t16, MROWS, 2 * CDIM, CDIM, 1);
  gemm(t16, w216, b2m, h32, pre3, nullptr, MROWS, CDIM, 2 * CDIM, 0);
  // norm3 + tanh -> out
  bn_stats_kernel<<<BATCH * CDIM, 256, 0, stream>>>(pre3, mean3, rstd3);
  bn_apply_kernel<<<(MROWS * CDIM + 255) / 256, 256, 0, stream>>>(
      pre3, mean3, rstd3, g3, b3, out, nullptr, MROWS * CDIM, 1);
}
